// LearnableMSNonLocalKernelDCT_86595130622085
// MI455X (gfx1250) — matrix operand order repaired and checked by reading, not compiled, not run on hardware
//
#include <hip/hip_runtime.h>

typedef __attribute__((ext_vector_type(16))) _Float16 v16h;
typedef __attribute__((ext_vector_type(8)))  _Float16 v8h;
typedef __attribute__((ext_vector_type(8)))  float    v8f;

// Problem shape (fixed by the reference setup_inputs()).
constexpr int B_ROWS = 1024;   // batch rows
constexpr int N_GRID = 4096;   // spatial grid / DCT size
constexpr int R_MS   = 256;    // active coefficients (k = 1..256)

constexpr float CSCALE    = 32.0f;              // folded into both C tables
constexpr float ASCALE    = 32.0f;              // extra scale on lam*a32
constexpr float OUT_SCALE = 1.0f / 32768.0f;    // undo 32 (gemm1) * 32 (scale) * 32 (gemm2 C)

// ---------------------------------------------------------------------------
// Precompute scaled DCT rows k=1..256 in two layouts, split into f16 hi/lo.
//   cb1 : [c][n]  (256 x 4096)  -> B operand of GEMM1 (contraction over n)
//   cb2 : [n][c]  (4096 x 256)  -> B operand of GEMM2 (contraction over c)
// Replicates the reference's float32 arithmetic: pi*(2n+1)*k/(2N), cosf.
// ---------------------------------------------------------------------------
__global__ void prep_c_kernel(_Float16* __restrict__ cb1_h, _Float16* __restrict__ cb1_l,
                              _Float16* __restrict__ cb2_h, _Float16* __restrict__ cb2_l) {
    int idx = blockIdx.x * blockDim.x + threadIdx.x;     // 0 .. 256*4096-1
    int c = idx >> 12;                                   // coefficient index 0..255  (k = c+1)
    int n = idx & (N_GRID - 1);
    float kf  = (float)(c + 1);
    float nf  = (float)n;
    float arg = ((3.14159265358979323846f * (2.0f * nf + 1.0f)) * kf) / (2.0f * (float)N_GRID);
    float val = CSCALE * 0.022097086912079612f /* sqrt(2/4096) */ * cosf(arg);
    _Float16 hi = (_Float16)val;
    _Float16 lo = (_Float16)(val - (float)hi);
    cb1_h[(size_t)c * N_GRID + n] = hi;
    cb1_l[(size_t)c * N_GRID + n] = lo;
    cb2_h[(size_t)n * R_MS + c]   = hi;
    cb2_l[(size_t)n * R_MS + c]   = lo;
}

// Split rho (fp32) into f16 hi/lo pair (Dekker split).
__global__ void split_rho_kernel(const float* __restrict__ rho,
                                 _Float16* __restrict__ rho_h, _Float16* __restrict__ rho_l) {
    int idx = blockIdx.x * blockDim.x + threadIdx.x;     // 0 .. 1024*4096-1
    float v = rho[idx];
    _Float16 hi = (_Float16)v;
    rho_h[idx] = hi;
    rho_l[idx] = (_Float16)(v - (float)hi);
}

// as = 32 * lam[c] * a32  (a32 = 32*a), split into f16 hi/lo.
__global__ void scale_a_kernel(const float* __restrict__ a32, const float* __restrict__ ms,
                               _Float16* __restrict__ as_h, _Float16* __restrict__ as_l) {
    int idx = blockIdx.x * blockDim.x + threadIdx.x;     // 0 .. 1024*256-1
    int c = idx & (R_MS - 1);
    float v = ASCALE * ms[c] * a32[idx];
    _Float16 hi = (_Float16)v;
    as_h[idx] = hi;
    as_l[idx] = (_Float16)(v - (float)hi);
}

// ---------------------------------------------------------------------------
// GEMM1: a32[1024 x 256] = rho[1024 x 4096] @ cb1^T    (K = 4096)
// One wave per 16x16 output tile; f16 hi/lo emulated-fp32 (hh + hl + lh).
// ---------------------------------------------------------------------------
__global__ void gemm1_kernel(const _Float16* __restrict__ rho_h, const _Float16* __restrict__ rho_l,
                             const _Float16* __restrict__ cb1_h, const _Float16* __restrict__ cb1_l,
                             float* __restrict__ a32) {
    const int lane = threadIdx.x & 31;
    const int wave = threadIdx.x >> 5;
    const int tile = blockIdx.x * 8 + wave;      // 1024 tiles total
    const int row_base = (tile >> 4) * 16;       // 64 row tiles
    const int col_base = (tile & 15) * 16;       // 16 col tiles
    const int idx16 = lane & 15;
    const int grp   = lane >> 4;

    const _Float16* aH = rho_h + (size_t)(row_base + idx16) * N_GRID;
    const _Float16* aL = rho_l + (size_t)(row_base + idx16) * N_GRID;
    const _Float16* bH = cb1_h + (size_t)(col_base + idx16) * N_GRID;
    const _Float16* bL = cb1_l + (size_t)(col_base + idx16) * N_GRID;

    v8f acc0 = {}; v8f acc1 = {}; v8f acc2 = {};
    for (int nb = 0; nb < N_GRID; nb += 32) {
        // A-fragment: lane (m=idx16, grp) holds K chunks [8g..8g+7], [16+8g..16+8g+7]
        v8h a0  = *(const v8h*)(aH + nb + 8 * grp);
        v8h a1  = *(const v8h*)(aH + nb + 16 + 8 * grp);
        v8h a0l = *(const v8h*)(aL + nb + 8 * grp);
        v8h a1l = *(const v8h*)(aL + nb + 16 + 8 * grp);
        // B-fragment: lane (n=idx16, grp) holds column n, K chunks [8g..8g+7], [16+8g..16+8g+7]
        v8h b0  = *(const v8h*)(bH + nb + 8 * grp);
        v8h b1  = *(const v8h*)(bH + nb + 16 + 8 * grp);
        v8h b0l = *(const v8h*)(bL + nb + 8 * grp);
        v8h b1l = *(const v8h*)(bL + nb + 16 + 8 * grp);
        __builtin_prefetch(bH + nb + 64, 0, 1);
        __builtin_prefetch(aH + nb + 64, 0, 1);
        v16h ah, al, bh, bl;
#pragma unroll
        for (int i = 0; i < 8; ++i) {
            ah[i] = a0[i];  ah[8 + i] = a1[i];
            al[i] = a0l[i]; al[8 + i] = a1l[i];
            bh[i] = b0[i];  bh[8 + i] = b1[i];
            bl[i] = b0l[i]; bl[8 + i] = b1l[i];
        }
        acc0 = __builtin_amdgcn_wmma_f32_16x16x32_f16(false, ah, false, bh, (short)0, acc0, false, false);
        acc1 = __builtin_amdgcn_wmma_f32_16x16x32_f16(false, ah, false, bl, (short)0, acc1, false, false);
        acc2 = __builtin_amdgcn_wmma_f32_16x16x32_f16(false, al, false, bh, (short)0, acc2, false, false);
    }
    v8f acc = acc0 + acc1 + acc2;
    // D layout: lane holds N = idx16, M = r + 8*grp
    float* out = a32 + (size_t)(row_base + 8 * grp) * R_MS + col_base + idx16;
#pragma unroll
    for (int r = 0; r < 8; ++r) out[(size_t)r * R_MS] = acc[r];
}

// ---------------------------------------------------------------------------
// GEMM2: phi[1024 x 4096] = as[1024 x 256] @ cb2^T-style   (K = 256)
// ---------------------------------------------------------------------------
__global__ void gemm2_kernel(const _Float16* __restrict__ as_h, const _Float16* __restrict__ as_l,
                             const _Float16* __restrict__ cb2_h, const _Float16* __restrict__ cb2_l,
                             float* __restrict__ phi) {
    const int lane = threadIdx.x & 31;
    const int wave = threadIdx.x >> 5;
    const int tile = blockIdx.x * 8 + wave;        // 16384 tiles total
    const int row_base = (tile >> 8) * 16;         // 64 row tiles
    const int col_base = (tile & 255) * 16;        // 256 col tiles
    const int idx16 = lane & 15;
    const int grp   = lane >> 4;

    const _Float16* aH = as_h  + (size_t)(row_base + idx16) * R_MS;
    const _Float16* aL = as_l  + (size_t)(row_base + idx16) * R_MS;
    const _Float16* bH = cb2_h + (size_t)(col_base + idx16) * R_MS;
    const _Float16* bL = cb2_l + (size_t)(col_base + idx16) * R_MS;

    v8f acc0 = {}; v8f acc1 = {}; v8f acc2 = {};
#pragma unroll
    for (int kb = 0; kb < R_MS; kb += 32) {
        v8h a0  = *(const v8h*)(aH + kb + 8 * grp);
        v8h a1  = *(const v8h*)(aH + kb + 16 + 8 * grp);
        v8h a0l = *(const v8h*)(aL + kb + 8 * grp);
        v8h a1l = *(const v8h*)(aL + kb + 16 + 8 * grp);
        v8h b0  = *(const v8h*)(bH + kb + 8 * grp);
        v8h b1  = *(const v8h*)(bH + kb + 16 + 8 * grp);
        v8h b0l = *(const v8h*)(bL + kb + 8 * grp);
        v8h b1l = *(const v8h*)(bL + kb + 16 + 8 * grp);
        v16h ah, al, bh, bl;
#pragma unroll
        for (int i = 0; i < 8; ++i) {
            ah[i] = a0[i];  ah[8 + i] = a1[i];
            al[i] = a0l[i]; al[8 + i] = a1l[i];
            bh[i] = b0[i];  bh[8 + i] = b1[i];
            bl[i] = b0l[i]; bl[8 + i] = b1l[i];
        }
        acc0 = __builtin_amdgcn_wmma_f32_16x16x32_f16(false, ah, false, bh, (short)0, acc0, false, false);
        acc1 = __builtin_amdgcn_wmma_f32_16x16x32_f16(false, ah, false, bl, (short)0, acc1, false, false);
        acc2 = __builtin_amdgcn_wmma_f32_16x16x32_f16(false, al, false, bh, (short)0, acc2, false, false);
    }
    v8f acc = (acc0 + acc1 + acc2);
    float* out = phi + (size_t)(row_base + 8 * grp) * N_GRID + col_base + idx16;
#pragma unroll
    for (int r = 0; r < 8; ++r) out[(size_t)r * N_GRID] = acc[r] * OUT_SCALE;
}

// ---------------------------------------------------------------------------
extern "C" void kernel_launch(void* const* d_in, const int* in_sizes, int n_in,
                              void* d_out, int out_size, void* d_ws, size_t ws_size,
                              hipStream_t stream) {
    const float* rho = (const float*)d_in[0];   // 1024*4096 f32
    const float* ms  = (const float*)d_in[1];   // 256 f32
    float* phi = (float*)d_out;                 // 1024*4096 f32

    char* ws = (char*)d_ws;
    size_t off = 0;
    _Float16* rho_h = (_Float16*)(ws + off); off += (size_t)B_ROWS * N_GRID * 2;   // 8 MB
    _Float16* rho_l = (_Float16*)(ws + off); off += (size_t)B_ROWS * N_GRID * 2;   // 8 MB
    _Float16* cb1_h = (_Float16*)(ws + off); off += (size_t)R_MS * N_GRID * 2;     // 2 MB
    _Float16* cb1_l = (_Float16*)(ws + off); off += (size_t)R_MS * N_GRID * 2;     // 2 MB
    _Float16* cb2_h = (_Float16*)(ws + off); off += (size_t)N_GRID * R_MS * 2;     // 2 MB
    _Float16* cb2_l = (_Float16*)(ws + off); off += (size_t)N_GRID * R_MS * 2;     // 2 MB
    float*    a32   = (float*)   (ws + off); off += (size_t)B_ROWS * R_MS * 4;     // 1 MB
    _Float16* as_h  = (_Float16*)(ws + off); off += (size_t)B_ROWS * R_MS * 2;     // 0.5 MB
    _Float16* as_l  = (_Float16*)(ws + off); off += (size_t)B_ROWS * R_MS * 2;     // 0.5 MB

    prep_c_kernel   <<<(R_MS * N_GRID) / 256,   256, 0, stream>>>(cb1_h, cb1_l, cb2_h, cb2_l);
    split_rho_kernel<<<(B_ROWS * N_GRID) / 256, 256, 0, stream>>>(rho, rho_h, rho_l);
    gemm1_kernel    <<<(B_ROWS / 16) * (R_MS / 16) / 8, 256, 0, stream>>>(rho_h, rho_l, cb1_h, cb1_l, a32);
    scale_a_kernel  <<<(B_ROWS * R_MS) / 256,   256, 0, stream>>>(a32, ms, as_h, as_l);
    gemm2_kernel    <<<(B_ROWS / 16) * (N_GRID / 16) / 8, 256, 0, stream>>>(as_h, as_l, cb2_h, cb2_l, phi);
}
